// HierarchicalFairnessMoE_37718402793542
// MI455X (gfx1250) — matrix operand order repaired and checked by reading, not compiled, not run on hardware
//
#include <hip/hip_runtime.h>
#include <hip/hip_bf16.h>

// ---------------------------------------------------------------------------
// HierarchicalFairnessMoE on MI455X (gfx1250, wave32, WMMA)
//
// BN folded into the first linear layers; dominant GEMMs run on
// v_wmma_f32_16x16x32_f16 (f32 accumulate). Expert pipeline fully fused:
// GEMM1 -> ReLU -> GEMM2 -> ReLU -> dot(w3) -> sigmoid -> router-weighted
// atomic accumulate; no [E,B,*] tensor ever hits HBM. Weights pre-transposed
// col-major so each lane's B fragment is two 16B reads. Each wave
// owns a 32x128 output tile (two A fragments) so every B fragment feeds two
// WMMAs, halving weight-fragment traffic through the WGP$.
// ---------------------------------------------------------------------------

#define MOE_B 8192
#define MOE_D 512
#define MOE_E 16
#define MOE_H 128
#define MOE_EPS 1e-5f

typedef __attribute__((ext_vector_type(16))) _Float16 v16h;
typedef __attribute__((ext_vector_type(8)))  _Float16 v8h;
typedef __attribute__((ext_vector_type(8)))  float    v8f;

__device__ __forceinline__ v16h cat16(v8h lo, v8h hi) {
    return __builtin_shufflevector(lo, hi, 0,1,2,3,4,5,6,7,8,9,10,11,12,13,14,15);
}

// ---------------------------------------------------------------------------
// Prep kernels
// ---------------------------------------------------------------------------

__global__ void moe_convert_x(const float* __restrict__ x, _Float16* __restrict__ xh) {
    int i = blockIdx.x * 256 + threadIdx.x;
    if (i < MOE_B * MOE_D) xh[i] = (_Float16)x[i];
}

// W1t[c][e][h][d] = (gamma*rsqrt(var+eps))[c,e,d] * eW1[c,e,d,h]   (col-major)
__global__ void moe_fuse_w1(const float* __restrict__ eW1,
                            const float* __restrict__ g,
                            const float* __restrict__ var,
                            _Float16* __restrict__ W1t) {
    int i = blockIdx.x * 256 + threadIdx.x;          // over 2*16*128*512
    if (i >= 2 * MOE_E * MOE_H * MOE_D) return;
    int d  = i & (MOE_D - 1);
    int h  = (i >> 9) & (MOE_H - 1);
    int ce = i >> 16;                                 // c*16+e
    float s = g[ce * MOE_D + d] * rsqrtf(var[ce * MOE_D + d] + MOE_EPS);
    W1t[i] = (_Float16)(s * eW1[((size_t)ce * MOE_D + d) * MOE_H + h]);
}

// b1f[c,e,h] = eb1 + sum_d (beta - mean*s) * eW1[c,e,d,h]
__global__ void moe_fuse_b1(const float* __restrict__ eW1,
                            const float* __restrict__ eb1,
                            const float* __restrict__ g,
                            const float* __restrict__ beta,
                            const float* __restrict__ mean,
                            const float* __restrict__ var,
                            float* __restrict__ b1f) {
    int i = blockIdx.x * 256 + threadIdx.x;          // over 2*16*128
    if (i >= 2 * MOE_E * MOE_H) return;
    int h  = i & (MOE_H - 1);
    int ce = i >> 7;
    float acc = eb1[i];
    for (int d = 0; d < MOE_D; ++d) {
        float s = g[ce * MOE_D + d] * rsqrtf(var[ce * MOE_D + d] + MOE_EPS);
        float t = beta[ce * MOE_D + d] - mean[ce * MOE_D + d] * s;
        acc += t * eW1[((size_t)ce * MOE_D + d) * MOE_H + h];
    }
    b1f[i] = acc;
}

// W2t[c][e][k][h] = eW2[c,e,h,k]   (col-major, f16)
__global__ void moe_w2t(const float* __restrict__ eW2, _Float16* __restrict__ W2t) {
    int i = blockIdx.x * 256 + threadIdx.x;          // over 2*16*64*128
    if (i >= 2 * MOE_E * 64 * MOE_H) return;
    int h  = i & (MOE_H - 1);
    int k  = (i >> 7) & 63;
    int ce = i >> 13;
    W2t[i] = (_Float16)eW2[((size_t)ce * MOE_H + h) * 64 + k];
}

__global__ void moe_zero_out(float* __restrict__ out) {
    int i = blockIdx.x * 256 + threadIdx.x;
    if (i < MOE_B) out[i] = 0.0f;
}

// ---------------------------------------------------------------------------
// Router: BN(folded) -> 512x64 -> ReLU -> 64x32 -> ReLU -> 32x16 -> softmax
// One thread per (branch, batch-row). ~1 GFLOP total, VALU is fine.
// ---------------------------------------------------------------------------
__global__ __launch_bounds__(128)
void moe_router(const float* __restrict__ x,
                const float* __restrict__ rbn_gamma, const float* __restrict__ rbn_beta,
                const float* __restrict__ rbn_mean,  const float* __restrict__ rbn_var,
                const float* __restrict__ rW1, const float* __restrict__ rb1,
                const float* __restrict__ rW2, const float* __restrict__ rb2,
                const float* __restrict__ rW3, const float* __restrict__ rb3,
                float* __restrict__ rw) {
    int b = blockIdx.x * 128 + threadIdx.x;
    int c = blockIdx.y;
    if (b >= MOE_B) return;

    const float* xr = x + (size_t)b * MOE_D;
    const float* gg = rbn_gamma + c * MOE_D;
    const float* bb = rbn_beta  + c * MOE_D;
    const float* mm = rbn_mean  + c * MOE_D;
    const float* vv = rbn_var   + c * MOE_D;

    float h1[64];
    #pragma unroll
    for (int j = 0; j < 64; ++j) h1[j] = rb1[c * 64 + j];
    for (int d = 0; d < MOE_D; ++d) {
        float s  = gg[d] * rsqrtf(vv[d] + MOE_EPS);
        float xn = (xr[d] - mm[d]) * s + bb[d];
        const float* w = rW1 + ((size_t)c * MOE_D + d) * 64;
        #pragma unroll
        for (int j = 0; j < 64; ++j) h1[j] = fmaf(xn, w[j], h1[j]);
    }
    #pragma unroll
    for (int j = 0; j < 64; ++j) h1[j] = h1[j] > 0.0f ? h1[j] : 0.0f;

    float h2[32];
    #pragma unroll
    for (int k = 0; k < 32; ++k) h2[k] = rb2[c * 32 + k];
    for (int j = 0; j < 64; ++j) {
        const float* w = rW2 + ((size_t)c * 64 + j) * 32;
        #pragma unroll
        for (int k = 0; k < 32; ++k) h2[k] = fmaf(h1[j], w[k], h2[k]);
    }
    #pragma unroll
    for (int k = 0; k < 32; ++k) h2[k] = h2[k] > 0.0f ? h2[k] : 0.0f;

    float lg[MOE_E];
    #pragma unroll
    for (int e = 0; e < MOE_E; ++e) lg[e] = rb3[c * MOE_E + e];
    for (int k = 0; k < 32; ++k) {
        const float* w = rW3 + ((size_t)c * 32 + k) * MOE_E;
        #pragma unroll
        for (int e = 0; e < MOE_E; ++e) lg[e] = fmaf(h2[k], w[e], lg[e]);
    }
    float mx = lg[0];
    #pragma unroll
    for (int e = 1; e < MOE_E; ++e) mx = fmaxf(mx, lg[e]);
    float sum = 0.0f;
    #pragma unroll
    for (int e = 0; e < MOE_E; ++e) { lg[e] = __expf(lg[e] - mx); sum += lg[e]; }
    float inv = 1.0f / sum;
    float* o = rw + ((size_t)c * MOE_B + b) * MOE_E;
    #pragma unroll
    for (int e = 0; e < MOE_E; ++e) o[e] = lg[e] * inv;
}

// ---------------------------------------------------------------------------
// Fused expert kernel. One workgroup = (128 batch rows) x (1 expert) x (1
// branch). 4 waves; wave w owns rows [32w, 32w+32) => two A fragments per
// k-step, each B fragment feeds two WMMAs.
// GEMM1: 16 k-steps x 8 col-tiles x 2 row-frags = 256 WMMA / wave.
// GEMM2: 4 k-steps x 4 col-tiles x 2 row-frags =  32 WMMA / wave
// (A re-shaped through a 32KB LDS h1 tile).
// ---------------------------------------------------------------------------
__global__ __launch_bounds__(128)
void moe_experts(const _Float16* __restrict__ xh,   // [B,512] row-major f16
                 const _Float16* __restrict__ W1t,  // [2,16,128,512] col-major
                 const float*    __restrict__ b1f,  // [2,16,128] fused bias
                 const _Float16* __restrict__ W2t,  // [2,16,64,128] col-major
                 const float*    __restrict__ eb2,  // [2,16,64]
                 const float*    __restrict__ eW3,  // [2,16,64]
                 const float*    __restrict__ eb3,  // [2,16]
                 const float*    __restrict__ rw,   // [2,8192,16]
                 const float*    __restrict__ bw,   // [2]
                 float* __restrict__ out) {         // [8192]
    __shared__ __align__(32) _Float16 h1s[128 * MOE_H];   // 32 KB

    const int tile = blockIdx.x;          // batch tile, 0..63
    const int e    = blockIdx.y;          // expert
    const int c    = blockIdx.z;          // branch
    const int ce   = c * MOE_E + e;
    const int lane = threadIdx.x & 31;
    const int wave = threadIdx.x >> 5;    // 0..3
    const int row0 = wave * 32;
    const int b0   = tile * 128;
    const int lo16 = lane & 15;
    const bool hi  = lane >= 16;
    const int rb   = hi ? 8 : 0;

    const _Float16* W1e = W1t + (size_t)ce * MOE_H * MOE_D;   // [h][d]
    const _Float16* W2e = W2t + (size_t)ce * 64 * MOE_H;      // [k][h]

    // ---- GEMM1: h1 = relu(x @ W1' + b1'), 32x128 per wave ----
    v8f acc[16];                          // acc[f*8+n]: row-frag f, col-tile n
    #pragma unroll
    for (int n = 0; n < 16; ++n) acc[n] = (v8f)(0.0f);

    const _Float16* aptr0 = xh + (size_t)(b0 + row0 +      lo16) * MOE_D;
    const _Float16* aptr1 = xh + (size_t)(b0 + row0 + 16 + lo16) * MOE_D;
    const int akb = hi ? 8 : 0;           // 16-bit A layout: lanes 16-31 hold K 8..15 / 24..31
    for (int k0 = 0; k0 < MOE_D; k0 += 32) {
        v16h af0 = cat16(*(const v8h*)(aptr0 + k0 + akb),
                         *(const v8h*)(aptr0 + k0 + akb + 16));
        v16h af1 = cat16(*(const v8h*)(aptr1 + k0 + akb),
                         *(const v8h*)(aptr1 + k0 + akb + 16));
        const int bkb = k0 + (hi ? 8 : 0);    // B layout: lanes 16-31 hold K 8..15 / 24..31
        #pragma unroll
        for (int n = 0; n < 8; ++n) {
            const _Float16* bp = W1e + (size_t)(n * 16 + lo16) * MOE_D + bkb;
            v16h bf = cat16(*(const v8h*)bp, *(const v8h*)(bp + 16));
            acc[n]     = __builtin_amdgcn_wmma_f32_16x16x32_f16(
                false, af0, false, bf, (short)0, acc[n],     false, false);
            acc[8 + n] = __builtin_amdgcn_wmma_f32_16x16x32_f16(
                false, af1, false, bf, (short)0, acc[8 + n], false, false);
        }
    }

    // bias + ReLU + stage h1 (f16) in LDS for GEMM2 fragment re-shape
    const float* b1e = b1f + ce * MOE_H;
    #pragma unroll
    for (int f = 0; f < 2; ++f) {
        #pragma unroll
        for (int n = 0; n < 8; ++n) {
            const int col = n * 16 + lo16;
            const float bias = b1e[col];
            #pragma unroll
            for (int r = 0; r < 8; ++r) {
                float v = acc[f * 8 + n][r] + bias;
                v = v > 0.0f ? v : 0.0f;
                h1s[(row0 + f * 16 + rb + r) * MOE_H + col] = (_Float16)v;
            }
        }
    }
    __syncthreads();

    // ---- GEMM2: h2 = relu(h1 @ W2 + b2), 32x64 per wave ----
    v8f acc2[8];                          // acc2[f*4+n]
    #pragma unroll
    for (int n = 0; n < 8; ++n) acc2[n] = (v8f)(0.0f);

    const _Float16* h1r0 = h1s + (row0 +      lo16) * MOE_H;
    const _Float16* h1r1 = h1s + (row0 + 16 + lo16) * MOE_H;
    for (int k0 = 0; k0 < MOE_H; k0 += 32) {
        v16h af0 = cat16(*(const v8h*)(h1r0 + k0 + akb),
                         *(const v8h*)(h1r0 + k0 + akb + 16));
        v16h af1 = cat16(*(const v8h*)(h1r1 + k0 + akb),
                         *(const v8h*)(h1r1 + k0 + akb + 16));
        const int bkb = k0 + (hi ? 8 : 0);
        #pragma unroll
        for (int n = 0; n < 4; ++n) {
            const _Float16* bp = W2e + (size_t)(n * 16 + lo16) * MOE_H + bkb;
            v16h bf = cat16(*(const v8h*)bp, *(const v8h*)(bp + 16));
            acc2[n]     = __builtin_amdgcn_wmma_f32_16x16x32_f16(
                false, af0, false, bf, (short)0, acc2[n],     false, false);
            acc2[4 + n] = __builtin_amdgcn_wmma_f32_16x16x32_f16(
                false, af1, false, bf, (short)0, acc2[4 + n], false, false);
        }
    }

    // ---- epilogue: relu(h2+b2) . w3, cross-lane reduce, sigmoid, accumulate ----
    const float* b2e = eb2 + ce * 64;
    const float* w3e = eW3 + ce * 64;
    const float  b3  = eb3[ce];

    float contrib[2][8];
    #pragma unroll
    for (int f = 0; f < 2; ++f)
        #pragma unroll
        for (int r = 0; r < 8; ++r) contrib[f][r] = 0.0f;

    #pragma unroll
    for (int f = 0; f < 2; ++f) {
        #pragma unroll
        for (int n = 0; n < 4; ++n) {
            const int col = n * 16 + lo16;
            const float bias = b2e[col];
            const float w3   = w3e[col];
            #pragma unroll
            for (int r = 0; r < 8; ++r) {
                float v = acc2[f * 4 + n][r] + bias;
                v = v > 0.0f ? v : 0.0f;
                contrib[f][r] = fmaf(v, w3, contrib[f][r]);
            }
        }
    }
    // reduce across the 16 column lanes of each half-wave
    #pragma unroll
    for (int f = 0; f < 2; ++f) {
        #pragma unroll
        for (int r = 0; r < 8; ++r) {
            float s = contrib[f][r];
            s += __shfl_xor(s, 1, 32);
            s += __shfl_xor(s, 2, 32);
            s += __shfl_xor(s, 4, 32);
            s += __shfl_xor(s, 8, 32);
            contrib[f][r] = s;
        }
    }
    if (lo16 < 8) {
        const int r = lo16;
        #pragma unroll
        for (int f = 0; f < 2; ++f) {
            const int gb = b0 + row0 + f * 16 + rb + r;
            const float dot = contrib[f][r] + b3;
            const float sg  = 1.0f / (1.0f + __expf(-dot));
            const float w   = bw[c] * rw[((size_t)c * MOE_B + gb) * MOE_E + e] * sg;
            atomicAdd(&out[gb], w);
        }
    }
}

// ---------------------------------------------------------------------------
// Launch
// ---------------------------------------------------------------------------
extern "C" void kernel_launch(void* const* d_in, const int* in_sizes, int n_in,
                              void* d_out, int out_size, void* d_ws, size_t ws_size,
                              hipStream_t stream) {
    const float* x         = (const float*)d_in[0];
    const float* rbn_gamma = (const float*)d_in[1];
    const float* rbn_beta  = (const float*)d_in[2];
    const float* rbn_mean  = (const float*)d_in[3];
    const float* rbn_var   = (const float*)d_in[4];
    const float* rW1 = (const float*)d_in[5];
    const float* rb1 = (const float*)d_in[6];
    const float* rW2 = (const float*)d_in[7];
    const float* rb2 = (const float*)d_in[8];
    const float* rW3 = (const float*)d_in[9];
    const float* rb3 = (const float*)d_in[10];
    const float* ebn_gamma = (const float*)d_in[11];
    const float* ebn_beta  = (const float*)d_in[12];
    const float* ebn_mean  = (const float*)d_in[13];
    const float* ebn_var   = (const float*)d_in[14];
    const float* eW1 = (const float*)d_in[15];
    const float* eb1 = (const float*)d_in[16];
    const float* eW2 = (const float*)d_in[17];
    const float* eb2 = (const float*)d_in[18];
    const float* eW3 = (const float*)d_in[19];
    const float* eb3 = (const float*)d_in[20];
    const float* bw  = (const float*)d_in[21];
    float* out = (float*)d_out;

    // workspace layout (bytes)
    char* ws = (char*)d_ws;
    _Float16* xh  = (_Float16*)(ws);                    //  8,388,608
    _Float16* W1t = (_Float16*)(ws + 8388608);          //  4,194,304
    _Float16* W2t = (_Float16*)(ws + 12582912);         //    524,288
    float*    b1f = (float*)   (ws + 13107200);         //     16,384
    float*    rwb = (float*)   (ws + 13123584);         //  1,048,576  (total ~14.2 MB)

    moe_convert_x<<<(MOE_B * MOE_D + 255) / 256, 256, 0, stream>>>(x, xh);
    moe_fuse_w1<<<(2 * MOE_E * MOE_H * MOE_D + 255) / 256, 256, 0, stream>>>(
        eW1, ebn_gamma, ebn_var, W1t);
    moe_fuse_b1<<<(2 * MOE_E * MOE_H + 255) / 256, 256, 0, stream>>>(
        eW1, eb1, ebn_gamma, ebn_beta, ebn_mean, ebn_var, b1f);
    moe_w2t<<<(2 * MOE_E * 64 * MOE_H + 255) / 256, 256, 0, stream>>>(eW2, W2t);
    moe_zero_out<<<(MOE_B + 255) / 256, 256, 0, stream>>>(out);
    moe_router<<<dim3(MOE_B / 128, 2), 128, 0, stream>>>(
        x, rbn_gamma, rbn_beta, rbn_mean, rbn_var,
        rW1, rb1, rW2, rb2, rW3, rb3, rwb);
    moe_experts<<<dim3(MOE_B / 128, MOE_E, 2), 128, 0, stream>>>(
        xh, W1t, b1f, W2t, eb2, eW3, eb3, rwb, bw, out);
}
